// ParamFreeCrossAttention_63702954934590
// MI455X (gfx1250) — matrix operand order repaired and checked by reading, not compiled, not run on hardware
//
#include <hip/hip_runtime.h>
#include <hip/hip_bf16.h>
#include <stdint.h>

typedef __attribute__((ext_vector_type(16))) __bf16 v16bf;
typedef __attribute__((ext_vector_type(8)))  __bf16 v8bf;
typedef __attribute__((ext_vector_type(8)))  float  v8f;
typedef __attribute__((ext_vector_type(4)))  unsigned int u32x4;
typedef __attribute__((ext_vector_type(8)))  int i32x8;
typedef __attribute__((ext_vector_type(4)))  int i32x4;

#define B_   4
#define H_   16
#define M_   1024
#define N_   2048
#define D_   1024
#define DH   64
#define NT   32            // N tile per iteration
#define NTILES (N_ / NT)   // 64
#define MTILE 256          // M rows per block (8 waves x 2 q-tiles x 16)
#define KSTR 72            // bf16 elems per K row in LDS (64 + 8 pad = 144B stride)
#define VSTR 40            // bf16 elems per V^T row in LDS (32 + 8 pad = 80B stride)
#define PSTR 40            // bf16 elems per P^T row in LDS
#define OSTR 36            // f32 elems per O staging row (32 + 4 pad)
#define LOG2E 1.44269504088896340736f

// Raw hardware ops (no denorm/precision fixup expansions)
__device__ __forceinline__ float fexp2(float x) { return __builtin_amdgcn_exp2f(x); }
__device__ __forceinline__ float frcp(float x)  { return __builtin_amdgcn_rcpf(x); }
__device__ __forceinline__ float frsq(float x)  { return __builtin_amdgcn_rsqf(x); }

__device__ __forceinline__ float silu_f(float x) {
  return x * frcp(1.0f + fexp2(-x * LOG2E));
}

__device__ __forceinline__ v8f wmma_bf16(v16bf a, v16bf b, v8f c) {
  // D = A(16x32 bf16) * B(32x16 bf16) + C(16x16 f32)
  return __builtin_amdgcn_wmma_f32_16x16x32_bf16(
      false, a, false, b, (short)0, c, false, false);
}

__device__ __forceinline__ v16bf cat8(v8bf lo, v8bf hi) {
  return __builtin_shufflevector(lo, hi,
      0, 1, 2, 3, 4, 5, 6, 7, 8, 9, 10, 11, 12, 13, 14, 15);
}

// ---------------------------------------------------------------------------
// TDM: async 2D tile load global->LDS (D# per CDNA5 ISA).
// data_size = 2 bytes; LDS padding inserted every 2^(padInt+1) dwords,
// (padAmt+1) dwords each, to match the padded LDS row strides.
// ---------------------------------------------------------------------------
__device__ __forceinline__ void tdm_load_2d(
    const void* gptr, unsigned int ldsAddr,
    unsigned int dim0, unsigned int dim1,       // tensor dims (elements)
    unsigned int tile0, unsigned int tile1,     // tile dims (elements)
    unsigned int stride0,                       // dim0 line stride (elements)
    unsigned int padInt, unsigned int padAmt) {
  const unsigned long long ga = (unsigned long long)(uintptr_t)gptr;
  u32x4 g0;
  g0[0] = 1u;                                            // count=1, user D#
  g0[1] = ldsAddr;                                       // lds_addr [63:32]
  g0[2] = (unsigned int)(ga & 0xFFFFFFFFu);              // global_addr lo
  g0[3] = (unsigned int)((ga >> 32) & 0x01FFFFFFu)       // global_addr hi
        | 0x80000000u;                                   // type=2 ("image")
  i32x8 g1;
  g1[0] = (int)((1u << 16)            // data_size = 2B
              | (1u << 20)            // pad_enable
              | (padInt << 22) | (padAmt << 25));
  g1[1] = (int)(dim0 << 16);                             // tensor_dim0 [79:48]
  g1[2] = (int)((dim0 >> 16) | (dim1 << 16));            // tensor_dim1 [111:80]
  g1[3] = (int)((dim1 >> 16) | (tile0 << 16));           // tile_dim0 [127:112]
  g1[4] = (int)(tile1 & 0xFFFFu);                        // tile_dim1 [143:128]
  g1[5] = (int)stride0;                                  // tensor_dim0_stride
  g1[6] = 0;
  g1[7] = 0;
  const i32x4 gz4 = {};
  const i32x8 gz8 = {};
  __builtin_amdgcn_tensor_load_to_lds(g0, g1, gz4, gz4, gz8, 0);
}

// ---------------------------------------------------------------------------
// Prep kernel: one pass over kv -> phi-normalized K (bf16, [bh][n][d]),
// mask-zeroed V transposed (bf16, [bh][d][n]), keep flags (bf16).
// ---------------------------------------------------------------------------
extern "C" __global__ __launch_bounds__(256, 1)
void pfca_prep(const float* __restrict__ kv, const uint8_t* __restrict__ kvmask,
               __bf16* __restrict__ wsK, __bf16* __restrict__ wsVT,
               __bf16* __restrict__ wsKeep) {
  const int tid = threadIdx.x;
  const int row = blockIdx.x * 32 + (tid >> 3);   // bh*N + n
  const int bh  = row >> 11;                      // / N_
  const int n   = row & (N_ - 1);
  const int b   = bh >> 4;
  const int h   = bh & 15;
  const int c0  = (tid & 7) * 8;

  const float* kr = kv + ((size_t)b * N_ + n) * D_ + (size_t)h * DH + c0;
  const float4 f0 = *(const float4*)(kr);
  const float4 f1 = *(const float4*)(kr + 4);
  const float x[8] = {f0.x, f0.y, f0.z, f0.w, f1.x, f1.y, f1.z, f1.w};
  float sl[8];
  float ss = 0.f;
  #pragma unroll
  for (int i = 0; i < 8; ++i) { sl[i] = silu_f(x[i]); ss += sl[i] * sl[i]; }
  ss += __shfl_xor(ss, 1, 32);    // 8 lanes share one kv row
  ss += __shfl_xor(ss, 2, 32);
  ss += __shfl_xor(ss, 4, 32);
  const float inv = fminf(frsq(ss), 1e6f);   // == 1/max(||x||, 1e-6)
  const float keepf = kvmask[(size_t)b * N_ + n] ? 0.0f : 1.0f;

  v8bf kx;
  #pragma unroll
  for (int i = 0; i < 8; ++i) kx[i] = (__bf16)(sl[i] * inv);
  *(v8bf*)(wsK + (size_t)row * DH + c0) = kx;
  #pragma unroll
  for (int i = 0; i < 8; ++i)    // one-time transposed scatter
    wsVT[((size_t)bh * DH + (c0 + i)) * N_ + n] = (__bf16)(x[i] * keepf);
  if ((tid & 7) == 0) wsKeep[(size_t)b * N_ + n] = (__bf16)keepf;
}

// ---------------------------------------------------------------------------
// Main flash-attention kernel. USEWS=true: TDM streams prepped bf16 tiles into
// LDS. USEWS=false: self-contained fallback (phi-norm computed in the fill).
// ---------------------------------------------------------------------------
template <bool USEWS>
__global__ __launch_bounds__(256, 1)
void pfca_kernel(const float* __restrict__ q, const float* __restrict__ kv,
                 const uint8_t* __restrict__ kvmask,
                 const __bf16* __restrict__ wsK, const __bf16* __restrict__ wsVT,
                 const __bf16* __restrict__ wsKeep, float* __restrict__ out) {
  __shared__ __bf16 ldsK[2][NT][KSTR];      // phi-normalized K tile [n][dh]
  __shared__ __bf16 ldsV[2][DH][VSTR];      // V^T tile [dh][n], masked cols zeroed
  __shared__ __bf16 ldsKeep[2][NT];         // keep flag (1/0) per kv row
  __shared__ __bf16 ldsP[8][2][16][PSTR];   // per-wave per-qtile P^T staging
  __shared__ float  ldsO[8][16][OSTR];      // per-wave O transpose staging

  const int tid  = threadIdx.x;
  const int lane = tid & 31;
  const int wave = tid >> 5;
  const int lrow = lane & 15;
  const int lgrp = lane >> 4;

  const int id = blockIdx.x;
  const int mt = id & 3;           // M_/MTILE = 4
  const int h  = (id >> 2) & 15;
  const int b  = id >> 6;
  const int mBase = mt * MTILE;

  // ---- Q (two 16-row tiles per wave): SiLU, L2-norm, fold log2e/sqrt(dh);
  // WMMA *B* operands: lane (col m=lrow) holds dh = hh*32 + lgrp*16 + [0..15]
  v16bf qb[2][2];
  #pragma unroll
  for (int qt = 0; qt < 2; ++qt) {
    const int mq = mBase + wave * 32 + qt * 16 + lrow;
    const float* qrow = q + ((size_t)b * M_ + mq) * D_ + (size_t)h * DH;
    float qv[2][16];
    float ssq = 0.f;
    #pragma unroll
    for (int hh = 0; hh < 2; ++hh) {
      const int off = hh * 32 + lgrp * 16;
      #pragma unroll
      for (int c = 0; c < 4; ++c) {
        const float4 f = *(const float4*)(qrow + off + c * 4);
        const float t[4] = {f.x, f.y, f.z, f.w};
        #pragma unroll
        for (int j = 0; j < 4; ++j) {
          const float s = silu_f(t[j]);
          ssq += s * s;
          qv[hh][c * 4 + j] = s;
        }
      }
    }
    ssq += __shfl_xor(ssq, 16, 32);   // lanes l, l+16 hold the row halves
    const float qscale = (0.125f * LOG2E) * fminf(frsq(ssq), 1e6f);
    #pragma unroll
    for (int hh = 0; hh < 2; ++hh)
      #pragma unroll
      for (int e = 0; e < 16; ++e)
        qb[qt][hh][e] = (__bf16)(qv[hh][e] * qscale);
  }

  // ---- K/V tile producer
  auto fillTile = [&](int buf, int nt) {
    if constexpr (USEWS) {
      if (wave == 0) {
        // Two async TDM tile loads; wave0 waits TENSORcnt before next barrier.
        tdm_load_2d(wsK + ((size_t)(b * H_ + h) * N_ + (size_t)nt * NT) * DH,
                    (unsigned int)(uintptr_t)&ldsK[buf][0][0],
                    /*dim0=*/DH, /*dim1=*/NT, /*tile0=*/DH, /*tile1=*/NT,
                    /*stride0=*/DH, /*padInt(32dw)=*/4, /*padAmt(4dw)=*/3);
        tdm_load_2d(wsVT + (size_t)(b * H_ + h) * DH * N_ + (size_t)nt * NT,
                    (unsigned int)(uintptr_t)&ldsV[buf][0][0],
                    /*dim0=*/NT, /*dim1=*/DH, /*tile0=*/NT, /*tile1=*/DH,
                    /*stride0=*/N_, /*padInt(16dw)=*/3, /*padAmt(4dw)=*/3);
        if (tid < 32)
          ldsKeep[buf][tid] = wsKeep[(size_t)b * N_ + (size_t)nt * NT + tid];
      }
    } else {
      const int r  = tid >> 3;        // kv row in tile
      const int c0 = (tid & 7) * 8;   // 8 contiguous dh elements
      const float* kr = kv + ((size_t)b * N_ + (size_t)nt * NT + r) * D_
                           + (size_t)h * DH + c0;
      if (nt + 1 < NTILES) __builtin_prefetch(kr + (size_t)NT * D_, 0, 1);
      const float keepf =
          kvmask[(size_t)b * N_ + (size_t)nt * NT + r] ? 0.0f : 1.0f;
      const float4 f0 = *(const float4*)(kr);
      const float4 f1 = *(const float4*)(kr + 4);
      const float x[8] = {f0.x, f0.y, f0.z, f0.w, f1.x, f1.y, f1.z, f1.w};
      float sl[8];
      float ss = 0.f;
      #pragma unroll
      for (int i = 0; i < 8; ++i) { sl[i] = silu_f(x[i]); ss += sl[i] * sl[i]; }
      ss += __shfl_xor(ss, 1, 32);
      ss += __shfl_xor(ss, 2, 32);
      ss += __shfl_xor(ss, 4, 32);
      const float inv = fminf(frsq(ss), 1e6f);
      #pragma unroll
      for (int i = 0; i < 8; ++i) ldsK[buf][r][c0 + i] = (__bf16)(sl[i] * inv);
      #pragma unroll
      for (int i = 0; i < 8; ++i)
        ldsV[buf][c0 + i][r] = (__bf16)(x[i] * keepf);
      if (tid < 32)
        ldsKeep[buf][tid] = (__bf16)(
            kvmask[(size_t)b * N_ + (size_t)nt * NT + tid] ? 0.0f : 1.0f);
    }
  };

  v8f oacc[2][4];   // O^T per q-tile; C-layout: lane col m=lrow, rows d=r+8*lgrp
  v8f lacc[2];      // softmax denominator accumulator (all rows identical)
  const v8f vzero = {};
  #pragma unroll
  for (int qt = 0; qt < 2; ++qt) {
    lacc[qt] = vzero;
    #pragma unroll
    for (int d = 0; d < 4; ++d) oacc[qt][d] = vzero;
  }

  fillTile(0, 0);

  for (int it = 0; it < NTILES; ++it) {
    const int buf = it & 1;
    if constexpr (USEWS) {
      if (wave == 0) __builtin_amdgcn_s_wait_tensorcnt(0);  // tile `it` landed
    }
    __syncthreads();
    if (it + 1 < NTILES) fillTile(buf ^ 1, it + 1);

    // ---- A operands shared by both q-tiles
    v16bf ak[2][2];   // K tile rows: [n-subtile][dh-half]
    #pragma unroll
    for (int ns = 0; ns < 2; ++ns) {
      const int n = ns * 16 + lrow;
      #pragma unroll
      for (int hh = 0; hh < 2; ++hh)
        ak[ns][hh] = cat8(
            *(const v8bf*)&ldsK[buf][n][hh * 32 + lgrp * 16],
            *(const v8bf*)&ldsK[buf][n][hh * 32 + lgrp * 16 + 8]);
    }
    const v16bf ka = cat8(*(const v8bf*)&ldsKeep[buf][lgrp * 8],
                          *(const v8bf*)&ldsKeep[buf][16 + lgrp * 8]);

    // ---- per q-tile: S^T = Kn*Qn^T, P = exp2(S^T), denominator via keep-WMMA
    v16bf pb[2];
    #pragma unroll
    for (int qt = 0; qt < 2; ++qt) {
      #pragma unroll
      for (int ns = 0; ns < 2; ++ns) {
        v8f st = vzero;
        st = wmma_bf16(ak[ns][0], qb[qt][0], st);
        st = wmma_bf16(ak[ns][1], qb[qt][1], st);
        v8bf pv;
        #pragma unroll
        for (int r = 0; r < 8; ++r)
          pv[r] = (__bf16)fexp2(st[r]);   // |st|<=0.19: bare v_exp_f32 is safe
        *(v8bf*)&ldsP[wave][qt][lrow][ns * 16 + 8 * lgrp] = pv;
      }
      // B operand: lane (col m=lrow) needs n = lgrp*8+[0..7], 16+lgrp*8+[0..7]; same-wave DS order
      pb[qt] = cat8(*(const v8bf*)&ldsP[wave][qt][lrow][lgrp * 8],
                    *(const v8bf*)&ldsP[wave][qt][lrow][16 + lgrp * 8]);
      lacc[qt] = wmma_bf16(ka, pb[qt], lacc[qt]);   // sum of kept p per row m
    }

    // ---- O^T += V^T * P^T : four d-subtiles, V operand reused across q-tiles
    #pragma unroll
    for (int d = 0; d < 4; ++d) {
      const int dr = d * 16 + lrow;
      const v16bf va = cat8(*(const v8bf*)&ldsV[buf][dr][lgrp * 8],
                            *(const v8bf*)&ldsV[buf][dr][16 + lgrp * 8]);
      #pragma unroll
      for (int qt = 0; qt < 2; ++qt)
        oacc[qt][d] = wmma_bf16(va, pb[qt], oacc[qt][d]);
    }
  }

  // ---- epilogue: normalize, transpose via wave-private LDS, coalesced stores
  #pragma unroll
  for (int qt = 0; qt < 2; ++qt) {
    const float invl = frcp(fmaxf((float)lacc[qt][0], 1e-30f));
    #pragma unroll
    for (int pass = 0; pass < 2; ++pass) {      // d in [pass*32, pass*32+32)
      #pragma unroll
      for (int dsub = 0; dsub < 2; ++dsub) {
        const int ds = pass * 2 + dsub;
        v8f sc;
        #pragma unroll
        for (int r = 0; r < 8; ++r) sc[r] = oacc[qt][ds][r] * invl;
        *(v8f*)&ldsO[wave][lrow][dsub * 16 + 8 * lgrp] = sc;
      }
      #pragma unroll
      for (int r2 = 0; r2 < 4; ++r2) {          // 4 rows x 8 lanes x float4
        const int mrow = r2 * 4 + (lane >> 3);
        const int c    = (lane & 7) * 4;
        const float4 vv = *(const float4*)&ldsO[wave][mrow][c];
        const int mg = mBase + wave * 32 + qt * 16 + mrow;
        *(float4*)(out + ((size_t)b * M_ + mg) * D_ + (size_t)h * DH
                   + pass * 32 + c) = vv;
      }
    }
  }
}

extern "C" void kernel_launch(void* const* d_in, const int* in_sizes, int n_in,
                              void* d_out, int out_size, void* d_ws, size_t ws_size,
                              hipStream_t stream) {
  (void)in_sizes; (void)n_in; (void)out_size;
  const float*   q      = (const float*)d_in[0];
  const float*   kv     = (const float*)d_in[1];
  const uint8_t* kvmask = (const uint8_t*)d_in[2];  // jax bool -> 1 byte/elem
  float*         out    = (float*)d_out;

  const size_t kvElems  = (size_t)B_ * H_ * N_ * DH;   // 8M
  const size_t needWs   = (2 * kvElems + (size_t)B_ * N_) * sizeof(unsigned short);

  dim3 grid(B_ * H_ * (M_ / MTILE));  // 256 workgroups
  dim3 block(256);                    // 8 wave32

  if (d_ws != nullptr && ws_size >= needWs) {
    __bf16* wsK    = (__bf16*)d_ws;
    __bf16* wsVT   = wsK + kvElems;
    __bf16* wsKeep = wsVT + kvElems;
    pfca_prep<<<dim3(B_ * H_ * N_ / 32), block, 0, stream>>>(
        kv, kvmask, wsK, wsVT, wsKeep);
    pfca_kernel<true><<<grid, block, 0, stream>>>(
        q, nullptr, nullptr, wsK, wsVT, wsKeep, out);
  } else {
    pfca_kernel<false><<<grid, block, 0, stream>>>(
        q, kv, kvmask, nullptr, nullptr, nullptr, out);
  }
}
